// TinyRecursiveModelTRMv3_55027120996914
// MI455X (gfx1250) — matrix operand order repaired and checked by reading, not compiled, not run on hardware
//
#include <hip/hip_runtime.h>
#include <math.h>

// ---------------------------------------------------------------------------
// CDNA5 (gfx1250) fully-fused TinyRecursiveModel, round 2.
// - MT=32 rows/block: halves L2 weight traffic per row vs MT=16.
// - Each wave computes BOTH 16-row M-tiles of a column strip from one B frag.
// - GELU fused into the WMMA epilogue (no f32 hidden buffer, fewer barriers).
// - 256 threads = 8 wave32s; all recurrent state in LDS (~218 KB < 320 KB/WGP).
// ---------------------------------------------------------------------------

typedef __attribute__((ext_vector_type(16))) _Float16 v16h;
typedef __attribute__((ext_vector_type(8)))  float    v8f;

#define MT    32      // rows per block
#define DD    128     // model dim
#define KMAX  5       // cached history steps (reference K=5)
#define NTHR  256     // 8 wave32s
#define SA    392     // f16 A-staging stride (halves, 16B multiple)
#define SH    264     // f16 hidden stride   (halves, 16B multiple)

// ---- dynamic LDS layout (byte offsets, all 16B aligned) ----
#define OFF_SX   0
#define OFF_SY   (OFF_SX + MT*DD*4)
#define OFF_SZ   (OFF_SY + MT*DD*4)
#define OFF_AF   (OFF_SZ + MT*DD*4)
#define OFF_Q    (OFF_AF + MT*DD*4)
#define OFF_O    (OFF_Q  + MT*DD*4)
#define OFF_AH   (OFF_O  + MT*DD*4)
#define OFF_H1H  (OFF_AH + MT*SA*2)
#define OFF_KC   (OFF_H1H + MT*SH*2)
#define OFF_VC   (OFF_KC + KMAX*MT*DD*2)
#define OFF_GATE (OFF_VC + KMAX*MT*DD*2)
#define OFF_PROB (OFF_GATE + MT*4)
#define SMEM_TOTAL (OFF_PROB + MT*8*4)     // 223360 bytes < 320KB/WGP

struct TrmParams {
  const float *x, *y0, *av;
  const float *llw, *llb, *law, *lab, *lzw, *lzb;
  const float *b1, *b2, *ba1, *ba2;
  const float *Wap, *bap, *Wag, *bag;
  const float *bq, *bk, *bv;
  const float *Wp1, *bp1, *Wp2, *bp2;
  const int   *Kp;
  const _Float16 *W1t, *W2t, *Wa1t, *Wa2t, *Wqt, *Wkt, *Wvt;  // transposed f16 [N][K]
  float *oy, *opad, *gsum;
  int B;
};

union Frag { uint4 u[2]; v16h v; };

// A fragment (16x32 f16, LDS row-major, stride sA halves):
// lane<16: M=lane, K = {kb..kb+7, kb+16..kb+23}; lane>=16: M=lane-16, K = {+8, +24}
__device__ __forceinline__ v16h load_a_frag(const _Float16* A, int sA, int kb, int lane) {
  const int hl = lane & 15, hi = lane >> 4;
  const _Float16* p = A + hl * sA + kb + hi * 8;
  Frag f;
  f.u[0] = *(const uint4*)(p);
  f.u[1] = *(const uint4*)(p + 16);
  return f.v;
}

// B fragment (32x16 f16) from transposed weights Bt[N][K]:
// lanes0-15: K={kb..kb+7, kb+16..kb+23} (col N=lane), lanes16-31: K={+8, +24} (col N=lane-16)
__device__ __forceinline__ v16h load_b_frag(const _Float16* Bt, int Kd, int ncol, int kb, int lane) {
  const int hl = lane & 15, hi = lane >> 4;
  const _Float16* p = Bt + (size_t)(ncol + hl) * Kd + kb + hi * 8;
  Frag f;
  f.u[0] = *(const uint4*)(p);
  f.u[1] = *(const uint4*)(p + 16);
  return f.v;
}

__device__ __forceinline__ float geluf(float x) {           // exact erf GELU
  return 0.5f * x * (1.0f + erff(x * 0.70710678118654752f));
}

enum EpiMode { EPI_F32 = 0, EPI_F16 = 1, EPI_GELU_F16 = 2 };

// C[32xNd] = A[32xKd] * W[KdxNd] + bias  (+ optional GELU / f16 narrow).
// Per column strip, one B fragment feeds both 16-row M-tiles (2 WMMAs).
// Nd is a multiple of 128 -> all 8 waves loop uniformly (EXEC all-ones at WMMA).
template <int MODE>
__device__ __forceinline__ void gemm2(const _Float16* A, int sA,
                                      const _Float16* Bt, int Kd, int Nd,
                                      void* Cv, int sC, const float* bias, int tid) {
  const int wave = tid >> 5, lane = tid & 31;
  const int hl = lane & 15, hi = lane >> 4;
  for (int nc = wave * 16; nc < Nd; nc += 8 * 16) {
    v8f acc0 = {}, acc1 = {};
    for (int kb = 0; kb < Kd; kb += 32) {
      v16h b  = load_b_frag(Bt, Kd, nc, kb, lane);
      v16h a0 = load_a_frag(A,            sA, kb, lane);   // rows 0..15
      v16h a1 = load_a_frag(A + 16 * sA,  sA, kb, lane);   // rows 16..31
      acc0 = __builtin_amdgcn_wmma_f32_16x16x32_f16(false, a0, false, b, (short)0, acc0, false, false);
      acc1 = __builtin_amdgcn_wmma_f32_16x16x32_f16(false, a1, false, b, (short)0, acc1, false, false);
    }
    const float bb = bias[nc + hl];
#pragma unroll
    for (int v = 0; v < 8; ++v) {                 // D layout: VGPR v -> M = v + 8*hi, N = nc + hl
      const int m0 = v + hi * 8;
      float r0 = acc0[v] + bb;
      float r1 = acc1[v] + bb;
      if (MODE == EPI_F32) {
        ((float*)Cv)[m0 * sC + nc + hl]        = r0;
        ((float*)Cv)[(16 + m0) * sC + nc + hl] = r1;
      } else if (MODE == EPI_F16) {
        ((_Float16*)Cv)[m0 * sC + nc + hl]        = (_Float16)r0;
        ((_Float16*)Cv)[(16 + m0) * sC + nc + hl] = (_Float16)r1;
      } else {
        ((_Float16*)Cv)[m0 * sC + nc + hl]        = (_Float16)geluf(r0);
        ((_Float16*)Cv)[(16 + m0) * sC + nc + hl] = (_Float16)geluf(r1);
      }
    }
  }
}

__device__ __forceinline__ float redsum8(float v) {         // sum across 8-lane group (wave32)
#pragma unroll
  for (int m = 4; m >= 1; m >>= 1) v += __shfl_xor(v, m, 8);
  return v;
}

__global__ __launch_bounds__(NTHR)
void trm_fused(TrmParams P) {
  extern __shared__ char smem[];
  float*    sx   = (float*)(smem + OFF_SX);
  float*    sy   = (float*)(smem + OFF_SY);
  float*    sz   = (float*)(smem + OFF_SZ);
  float*    af   = (float*)(smem + OFF_AF);
  float*    qb   = (float*)(smem + OFF_Q);
  float*    ob   = (float*)(smem + OFF_O);
  _Float16* ah   = (_Float16*)(smem + OFF_AH);
  _Float16* h1h  = (_Float16*)(smem + OFF_H1H);
  _Float16* kc   = (_Float16*)(smem + OFF_KC);
  _Float16* vc   = (_Float16*)(smem + OFF_VC);
  float*    gate = (float*)(smem + OFF_GATE);
  float*    prob = (float*)(smem + OFF_PROB);

  const int tid = threadIdx.x;
  const int r0  = blockIdx.x * MT;
  const int Kv  = (P.Kp[0] < KMAX) ? P.Kp[0] : KMAX;
  const int row = tid >> 3, jj = tid & 7;      // 8 threads per row for LN/reductions

  if (tid == 0) {                              // warm L2 path for the weight stream
    __builtin_prefetch(P.W1t, 0, 1);
    __builtin_prefetch(P.Wa1t, 0, 1);
  }

  // ---- stage x, y; z=0; affect latent (constant over iterations) ----
  for (int p = tid; p < MT * DD; p += NTHR) {
    int m = p >> 7, d = p & 127;
    sx[p] = P.x [(size_t)(r0 + m) * DD + d];
    sy[p] = P.y0[(size_t)(r0 + m) * DD + d];
    sz[p] = 0.0f;
    float a0 = P.av[(r0 + m) * 3 + 0];
    float a1 = P.av[(r0 + m) * 3 + 1];
    float a2 = P.av[(r0 + m) * 3 + 2];
    af[p] = tanhf(a0 * P.Wap[d] + a1 * P.Wap[DD + d] + a2 * P.Wap[2 * DD + d] + P.bap[d]);
  }
  __syncthreads();

  for (int k = 0; k < Kv; ++k) {
    // ============ INNER latent recurrence (x4) ============
    for (int it = 0; it < 4; ++it) {
      // LN over concat(x,y,z) [384] -> f16 A staging (two-pass, 8 thr/row)
      {
        float s1 = 0.f, s2 = 0.f;
        for (int i = 0; i < 48; ++i) {
          int f = jj * 48 + i;
          float v = (f < 128) ? sx[row * DD + f]
                  : (f < 256) ? sy[row * DD + f - 128]
                              : sz[row * DD + f - 256];
          s1 += v; s2 += v * v;
        }
        s1 = redsum8(s1); s2 = redsum8(s2);
        float mean = s1 * (1.f / 384.f);
        float rstd = rsqrtf(s2 * (1.f / 384.f) - mean * mean + 1e-5f);
        for (int i = 0; i < 48; ++i) {
          int f = jj * 48 + i;
          float v = (f < 128) ? sx[row * DD + f]
                  : (f < 256) ? sy[row * DD + f - 128]
                              : sz[row * DD + f - 256];
          ah[row * SA + f] = (_Float16)((v - mean) * rstd * P.llw[f] + P.llb[f]);
        }
      }
      __syncthreads();

      gemm2<EPI_GELU_F16>(ah, SA, P.W1t, 384, 256, h1h, SH, P.b1, tid);
      __syncthreads();
      gemm2<EPI_F32>(h1h, SH, P.W2t, 256, 128, ob, DD, P.b2, tid);
      __syncthreads();

      // z = LN(z + 0.3 * z_delta)   (two-pass)
      {
        float t1 = 0.f, t2 = 0.f;
        for (int i = 0; i < 16; ++i) {
          int f = jj * 16 + i;
          float v = sz[row * DD + f] + 0.3f * ob[row * DD + f];
          t1 += v; t2 += v * v;
        }
        t1 = redsum8(t1); t2 = redsum8(t2);
        float mean = t1 * (1.f / 128.f);
        float rstd = rsqrtf(t2 * (1.f / 128.f) - mean * mean + 1e-5f);
        for (int i = 0; i < 16; ++i) {
          int f = jj * 16 + i;
          float v = sz[row * DD + f] + 0.3f * ob[row * DD + f];
          sz[row * DD + f] = (v - mean) * rstd * P.lzw[f] + P.lzb[f];
        }
      }
      __syncthreads();
    }

    // ============ affect gate ============
    {
      float t = 0.f;
      for (int i = 0; i < 16; ++i) {
        int f = jj * 16 + i;
        t += sz[row * DD + f] * P.Wag[f] + af[row * DD + f] * P.Wag[DD + f];
      }
      t = redsum8(t);
      if (jj == 0) gate[row] = 1.f / (1.f + expf(-(t + P.bag[0])));
    }
    __syncthreads();
    if (tid == 0) {                               // deterministic conf accumulation
      float sg = 0.f;
      for (int m = 0; m < MT; ++m) sg += gate[m];
      P.gsum[(size_t)k * gridDim.x + blockIdx.x] = sg;
    }
    for (int p = tid; p < MT * DD; p += NTHR) {   // z += 0.3*gate*affect; stage z as f16
      int m = p >> 7;
      float z = sz[p] + 0.3f * gate[m] * af[p];
      sz[p] = z;
      ah[m * SA + (p & 127)] = (_Float16)z;
    }
    __syncthreads();

    // ============ history append: q (current), k_t/v_t cached ============
    gemm2<EPI_F32>(ah, SA, P.Wqt, 128, 128, qb, DD, P.bq, tid);
    gemm2<EPI_F16>(ah, SA, P.Wkt, 128, 128, kc + k * MT * DD, DD, P.bk, tid);
    gemm2<EPI_F16>(ah, SA, P.Wvt, 128, 128, vc + k * MT * DD, DD, P.bv, tid);
    __syncthreads();

    // ============ attention over history (last query only) ============
    if (k > 0) {
      const int t = k + 1;
      if (tid < MT * t) {
        int m = tid / t, s = tid % t;
        float acc = 0.f;
        for (int d = 0; d < DD; ++d)
          acc += qb[m * DD + d] * (float)kc[(s * MT + m) * DD + d];
        prob[m * 8 + s] = acc * 0.088388347648318447f;   // 1/sqrt(128)
      }
      __syncthreads();
      if (tid < MT) {
        float mx = -1e30f;
        for (int s = 0; s < t; ++s) mx = fmaxf(mx, prob[tid * 8 + s]);
        float ssum = 0.f;
        for (int s = 0; s < t; ++s) {
          float e = expf(prob[tid * 8 + s] - mx);
          prob[tid * 8 + s] = e; ssum += e;
        }
        float inv = 1.f / ssum;
        for (int s = 0; s < t; ++s) prob[tid * 8 + s] *= inv;
      }
      __syncthreads();
      for (int p = tid; p < MT * DD; p += NTHR) {
        int m = p >> 7, d = p & 127;
        float a = 0.f;
        for (int s = 0; s < t; ++s)
          a += prob[m * 8 + s] * (float)vc[(s * MT + m) * DD + d];
        ob[p] = sz[p] + a;
      }
      __syncthreads();
      {                                            // z = LN(z + attn)
        float u1 = 0.f, u2 = 0.f;
        for (int i = 0; i < 16; ++i) {
          int f = jj * 16 + i;
          float v = ob[row * DD + f];
          u1 += v; u2 += v * v;
        }
        u1 = redsum8(u1); u2 = redsum8(u2);
        float mean = u1 * (1.f / 128.f);
        float rstd = rsqrtf(u2 * (1.f / 128.f) - mean * mean + 1e-5f);
        for (int i = 0; i < 16; ++i) {
          int f = jj * 16 + i;
          sz[row * DD + f] = (ob[row * DD + f] - mean) * rstd * P.lzw[f] + P.lzb[f];
        }
      }
      __syncthreads();
    }

    // ============ answer MLP ============
    {
      float w1 = 0.f, w2 = 0.f;                    // LN over concat(y,z) [256]
      for (int i = 0; i < 32; ++i) {
        int f = jj * 32 + i;
        float v = (f < 128) ? sy[row * DD + f] : sz[row * DD + f - 128];
        w1 += v; w2 += v * v;
      }
      w1 = redsum8(w1); w2 = redsum8(w2);
      float meanA = w1 * (1.f / 256.f);
      float rstdA = rsqrtf(w2 * (1.f / 256.f) - meanA * meanA + 1e-5f);
      for (int i = 0; i < 32; ++i) {
        int f = jj * 32 + i;
        float v = (f < 128) ? sy[row * DD + f] : sz[row * DD + f - 128];
        ah[row * SA + f] = (_Float16)((v - meanA) * rstdA * P.law[f] + P.lab[f]);
      }
    }
    __syncthreads();
    gemm2<EPI_GELU_F16>(ah, SA, P.Wa1t, 256, 256, h1h, SH, P.ba1, tid);
    __syncthreads();
    gemm2<EPI_F32>(h1h, SH, P.Wa2t, 256, 128, ob, DD, P.ba2, tid);
    __syncthreads();
    for (int p = tid; p < MT * DD; p += NTHR) sy[p] += 0.4f * ob[p];
    __syncthreads();
  }

  // ============ outputs: y, pad head ============
  for (int p = tid; p < MT * DD; p += NTHR) {
    int m = p >> 7, d = p & 127;
    P.oy[(size_t)(r0 + m) * DD + d] = sy[p];
  }
  for (int p = tid; p < MT * 64; p += NTHR) {      // h = gelu(z @ W_p1 + b_p1)
    int m = p >> 6, c = p & 63;
    float s = P.bp1[c];
    for (int d = 0; d < DD; ++d) s += sz[m * DD + d] * P.Wp1[d * 64 + c];
    qb[m * 64 + c] = geluf(s);                     // qb reused as scratch
  }
  __syncthreads();
  if (tid < MT * 3) {                              // pad = tanh(h @ W_p2 + b_p2)
    int m = tid / 3, c = tid % 3;
    float s = P.bp2[c];
    for (int j2 = 0; j2 < 64; ++j2) s += qb[m * 64 + j2] * P.Wp2[j2 * 3 + c];
    P.opad[(size_t)(r0 + m) * 3 + c] = tanhf(s);
  }
}

// ---- setup: fp32 [K][N] -> f16 transposed [N][K] ----
__global__ void cvt_transpose(const float* __restrict__ src, _Float16* __restrict__ dst,
                              int K, int N) {
  int i = blockIdx.x * blockDim.x + threadIdx.x;
  if (i < K * N) {
    int k = i / N, n = i % N;
    dst[(size_t)n * K + k] = (_Float16)src[i];
  }
}

// ---- deterministic conf reduction ----
__global__ void conf_reduce(const float* __restrict__ gsum, float* __restrict__ oconf,
                            int nb, int Kout, float invB) {
  int k = threadIdx.x;
  if (k < Kout) {
    float s = 0.f;
    for (int i = 0; i < nb; ++i) s += gsum[(size_t)k * nb + i];
    oconf[k] = s * invB;
  }
}

extern "C" void kernel_launch(void* const* d_in, const int* in_sizes, int n_in,
                              void* d_out, int out_size, void* d_ws, size_t ws_size,
                              hipStream_t stream) {
  (void)n_in; (void)ws_size;
  const int B    = in_sizes[0] / DD;          // 32768
  const int nb   = B / MT;                    // 1024 blocks
  const int Kout = out_size - B * (DD + 3);   // number of conf entries (=K)

  // workspace carve: transposed f16 weights, then per-block gate sums
  _Float16* w = (_Float16*)d_ws;
  size_t off = 0;
  _Float16* W1t  = w + off; off += 384 * 256;
  _Float16* W2t  = w + off; off += 256 * 128;
  _Float16* Wa1t = w + off; off += 256 * 256;
  _Float16* Wa2t = w + off; off += 256 * 128;
  _Float16* Wqt  = w + off; off += 128 * 128;
  _Float16* Wkt  = w + off; off += 128 * 128;
  _Float16* Wvt  = w + off; off += 128 * 128;
  float* gsum = (float*)(w + off);            // KMAX * nb floats

  cvt_transpose<<<(384*256 + 255) / 256, 256, 0, stream>>>((const float*)d_in[9],  W1t, 384, 256);
  cvt_transpose<<<(256*128 + 255) / 256, 256, 0, stream>>>((const float*)d_in[11], W2t, 256, 128);
  cvt_transpose<<<(256*256 + 255) / 256, 256, 0, stream>>>((const float*)d_in[13], Wa1t, 256, 256);
  cvt_transpose<<<(256*128 + 255) / 256, 256, 0, stream>>>((const float*)d_in[15], Wa2t, 256, 128);
  cvt_transpose<<<(128*128 + 255) / 256, 256, 0, stream>>>((const float*)d_in[21], Wqt, 128, 128);
  cvt_transpose<<<(128*128 + 255) / 256, 256, 0, stream>>>((const float*)d_in[23], Wkt, 128, 128);
  cvt_transpose<<<(128*128 + 255) / 256, 256, 0, stream>>>((const float*)d_in[25], Wvt, 128, 128);

  float* oy    = (float*)d_out;
  float* oconf = oy + (size_t)B * DD;
  float* opad  = oconf + Kout;

  TrmParams P;
  P.x   = (const float*)d_in[0];  P.y0  = (const float*)d_in[1];  P.av  = (const float*)d_in[2];
  P.llw = (const float*)d_in[3];  P.llb = (const float*)d_in[4];
  P.law = (const float*)d_in[5];  P.lab = (const float*)d_in[6];
  P.lzw = (const float*)d_in[7];  P.lzb = (const float*)d_in[8];
  P.b1  = (const float*)d_in[10]; P.b2  = (const float*)d_in[12];
  P.ba1 = (const float*)d_in[14]; P.ba2 = (const float*)d_in[16];
  P.Wap = (const float*)d_in[17]; P.bap = (const float*)d_in[18];
  P.Wag = (const float*)d_in[19]; P.bag = (const float*)d_in[20];
  P.bq  = (const float*)d_in[22]; P.bk  = (const float*)d_in[24]; P.bv = (const float*)d_in[26];
  P.Wp1 = (const float*)d_in[27]; P.bp1 = (const float*)d_in[28];
  P.Wp2 = (const float*)d_in[29]; P.bp2 = (const float*)d_in[30];
  P.Kp  = (const int*)d_in[31];
  P.W1t = W1t; P.W2t = W2t; P.Wa1t = Wa1t; P.Wa2t = Wa2t;
  P.Wqt = Wqt; P.Wkt = Wkt; P.Wvt = Wvt;
  P.oy = oy; P.opad = opad; P.gsum = gsum; P.B = B;

  trm_fused<<<nb, NTHR, SMEM_TOTAL, stream>>>(P);
  conf_reduce<<<1, 32, 0, stream>>>(gsum, oconf, nb, Kout, 1.0f / (float)B);
}
